// GAT_33174327394865
// MI455X (gfx1250) — matrix operand order repaired and checked by reading, not compiled, not run on hardware
//
#include <hip/hip_runtime.h>
#include <hip/hip_bf16.h>

#define B_   4
#define N_   2048
#define FIN  256
#define FOUT 256
#define H_   4
#define LEAKY 0.2f
#define NT_  (N_ / 32)   // 64 attention K-tiles

typedef __attribute__((ext_vector_type(16))) _Float16 v16h;
typedef __attribute__((ext_vector_type(8)))  float    v8f;
typedef __attribute__((address_space(3))) float lds_float;

union AFrag { v16h v; unsigned int u[8]; _Float16 h[16]; };
union H8    { uint4 u4; _Float16 h[8]; };
union H2    { unsigned int u; _Float16 h[2]; };

__device__ inline v8f wmma_f16(v16h a, v16h b, v8f c) {
  return __builtin_amdgcn_wmma_f32_16x16x32_f16(false, a, false, b, (short)0, c,
                                                false, false);
}

__device__ inline void load_frag32B(AFrag& f, const _Float16* p) {
  uint4 r0 = reinterpret_cast<const uint4*>(p)[0];
  uint4 r1 = reinterpret_cast<const uint4*>(p)[2];
  f.u[0] = r0.x; f.u[1] = r0.y; f.u[2] = r0.z; f.u[3] = r0.w;
  f.u[4] = r1.x; f.u[5] = r1.y; f.u[6] = r1.z; f.u[7] = r1.w;
}

// ---- async global->LDS staging (CDNA5 GLOBAL_LOAD_ASYNC_TO_LDS_B128) ------
// Emitted via inline asm: vdst = 32-bit LDS byte offset, vaddr = 64-bit
// per-lane global address. Tracked by ASYNCcnt, waited with s_wait_asynccnt.
__device__ inline void bias_prefetch(const float* g, float* l) {
  unsigned loff = (unsigned)(unsigned long long)(lds_float*)l;
  unsigned long long ga = (unsigned long long)g;
  asm volatile("global_load_async_to_lds_b128 %0, %1, off"
               :: "v"(loff), "v"(ga) : "memory");
}
__device__ inline void bias_wait(bool one_outstanding) {
  if (one_outstanding)
    asm volatile("s_wait_asynccnt 0x1" ::: "memory");
  else
    asm volatile("s_wait_asynccnt 0x0" ::: "memory");
}

// ---------------------------------------------------------------- converts --
__global__ void k_cvt_x(const float* __restrict__ x, _Float16* __restrict__ x16,
                        int n) {
  int i = blockIdx.x * blockDim.x + threadIdx.x;
  if (i < n) x16[i] = (_Float16)x[i];
}

// W [H][FIN][FOUT] f32 -> W16T [H][FOUT][FIN] f16
__global__ void k_cvt_wt(const float* __restrict__ w, _Float16* __restrict__ wt) {
  int idx = blockIdx.x * blockDim.x + threadIdx.x;
  if (idx >= H_ * FOUT * FIN) return;
  int h = idx / (FOUT * FIN);
  int r = idx % (FOUT * FIN);
  int o = r / FIN;
  int f = r % FIN;
  wt[idx] = (_Float16)w[(h * FIN + f) * FOUT + o];
}

// ------------------------------------------------------------------- GEMM1 --
// T = X @ W_h + b_h ; stored transposed as T16T[h][b][o][n] (f16)
// Two-stage register pipeline over kk (fully unrolled -> constant indices).
__global__ __launch_bounds__(128) void k_gemm1(
    const _Float16* __restrict__ x16, const _Float16* __restrict__ wt,
    const float* __restrict__ bvec, _Float16* __restrict__ t16t) {
  const int hb = blockIdx.y;
  const int h = hb / B_, bb = hb % B_;
  const int i0 = blockIdx.x * 16;
  const int lane = threadIdx.x & 31;
  const int wave = threadIdx.x >> 5;
  const int hi = lane >> 4, lo = lane & 15;
  const int obase = wave * 64;

  v8f acc[4] = {};
  const _Float16* xrow = x16 + ((size_t)(bb * N_ + i0 + lo)) * FIN + hi * 8;
  const _Float16* wbase[4];
#pragma unroll
  for (int t = 0; t < 4; ++t)
    wbase[t] = wt + ((size_t)(h * FOUT + obase + t * 16 + lo)) * FIN + hi * 8;

  AFrag a[2];
  AFrag bf[2][4];
  // prologue: stage kk = 0
  {
    uint4 q0 = *reinterpret_cast<const uint4*>(xrow);
    uint4 q1 = *reinterpret_cast<const uint4*>(xrow + 16);
    a[0].u[0] = q0.x; a[0].u[1] = q0.y; a[0].u[2] = q0.z; a[0].u[3] = q0.w;
    a[0].u[4] = q1.x; a[0].u[5] = q1.y; a[0].u[6] = q1.z; a[0].u[7] = q1.w;
#pragma unroll
    for (int t = 0; t < 4; ++t) load_frag32B(bf[0][t], wbase[t]);
  }
#pragma unroll
  for (int kk = 0; kk < FIN / 32; ++kk) {
    const int cur = kk & 1, nxt = cur ^ 1;
    if (kk + 1 < FIN / 32) {  // prefetch next K-slab while current computes
      uint4 q0 = *reinterpret_cast<const uint4*>(xrow + (kk + 1) * 32);
      uint4 q1 = *reinterpret_cast<const uint4*>(xrow + (kk + 1) * 32 + 16);
      a[nxt].u[0] = q0.x; a[nxt].u[1] = q0.y; a[nxt].u[2] = q0.z; a[nxt].u[3] = q0.w;
      a[nxt].u[4] = q1.x; a[nxt].u[5] = q1.y; a[nxt].u[6] = q1.z; a[nxt].u[7] = q1.w;
#pragma unroll
      for (int t = 0; t < 4; ++t)
        load_frag32B(bf[nxt][t], wbase[t] + (kk + 1) * 32);
    }
#pragma unroll
    for (int t = 0; t < 4; ++t)
      acc[t] = wmma_f16(a[cur].v, bf[cur][t].v, acc[t]);
  }
#pragma unroll
  for (int t = 0; t < 4; ++t) {
    int o = obase + t * 16 + lo;
    float bo = bvec[h * FOUT + o];
    H8 pk;
#pragma unroll
    for (int r = 0; r < 8; ++r) pk.h[r] = (_Float16)(acc[t][r] + bo);
    _Float16* dst = t16t + ((size_t)(hb * FOUT + o)) * N_ + i0 + hi * 8;
    *reinterpret_cast<uint4*>(dst) = pk.u4;
  }
}

// ------------------------------------------------------------------- score --
__global__ __launch_bounds__(256) void k_score(
    const _Float16* __restrict__ t16t, const float* __restrict__ av,
    float* __restrict__ s) {
  int hb = blockIdx.y;
  int h = hb / B_;
  int n = blockIdx.x * 256 + threadIdx.x;
  const _Float16* base = t16t + (size_t)hb * FOUT * N_ + n;
  const float* a = av + h * FOUT;
  float acc = 0.f;
  for (int o = 0; o < FOUT; ++o) acc += (float)base[(size_t)o * N_] * a[o];
  s[hb * N_ + n] = acc;
}

// ------------------------------------------------------------------- stats --
__global__ __launch_bounds__(256) void k_stats(
    const float* __restrict__ bias, const float* __restrict__ s,
    float* __restrict__ rmax, float* __restrict__ rinv) {
  const int i = blockIdx.x, bb = blockIdx.y, tid = threadIdx.x;
  const float* brow = bias + ((size_t)bb * N_ + i) * N_;
  float si[H_], ml[H_], ll[H_];
#pragma unroll
  for (int h = 0; h < H_; ++h) {
    si[h] = s[(h * B_ + bb) * N_ + i];
    ml[h] = -3.4e38f;
    ll[h] = 0.f;
  }
  for (int j = tid; j < N_; j += 256) {
    float bv = brow[j];
#pragma unroll
    for (int h = 0; h < H_; ++h) {
      float x = si[h] + s[(h * B_ + bb) * N_ + j];
      x = x > 0.f ? x : LEAKY * x;
      x += bv;
      ml[h] = fmaxf(ml[h], x);
    }
  }
  __shared__ float red[H_][256];
#pragma unroll
  for (int h = 0; h < H_; ++h) red[h][tid] = ml[h];
  __syncthreads();
  for (int st = 128; st > 0; st >>= 1) {
    if (tid < st) {
#pragma unroll
      for (int h = 0; h < H_; ++h)
        red[h][tid] = fmaxf(red[h][tid], red[h][tid + st]);
    }
    __syncthreads();
  }
  float M[H_];
#pragma unroll
  for (int h = 0; h < H_; ++h) M[h] = red[h][0];
  __syncthreads();
  for (int j = tid; j < N_; j += 256) {
    float bv = brow[j];
#pragma unroll
    for (int h = 0; h < H_; ++h) {
      float x = si[h] + s[(h * B_ + bb) * N_ + j];
      x = x > 0.f ? x : LEAKY * x;
      x += bv;
      ll[h] += __expf(x - M[h]);
    }
  }
#pragma unroll
  for (int h = 0; h < H_; ++h) red[h][tid] = ll[h];
  __syncthreads();
  for (int st = 128; st > 0; st >>= 1) {
    if (tid < st) {
#pragma unroll
      for (int h = 0; h < H_; ++h) red[h][tid] += red[h][tid + st];
    }
    __syncthreads();
  }
  if (tid == 0) {
#pragma unroll
    for (int h = 0; h < H_; ++h) {
      rmax[((size_t)h * B_ + bb) * N_ + i] = M[h];
      rinv[((size_t)h * B_ + bb) * N_ + i] = 1.f / red[h][0];
    }
  }
}

// -------------------------------------------------------------- attn GEMM2 --
// out[b][i][o] = (1/H) * sum_h sum_j P_h[i][j] * T_h[j][o].
// bias tiles: async global->LDS double buffer (ASYNCcnt);
// T fragments: register double buffer (jt unrolled by 2).
__global__ __launch_bounds__(128) void k_attn(
    const float* __restrict__ bias, const float* __restrict__ s,
    const float* __restrict__ rmax, const float* __restrict__ rinv,
    const _Float16* __restrict__ t16t, float* __restrict__ out) {
  __shared__ unsigned int plds[16 * 32 / 2];            // 16x32 f16 P tile
  __shared__ __align__(16) float bias_lds[2][16 * 32];  // double-buffered bias
  __shared__ float row_s[16], row_m[16], row_il[16];

  const int bb = blockIdx.y;
  const int i0 = blockIdx.x * 16;
  const int tid = threadIdx.x;
  const int lane = tid & 31, wave = tid >> 5;
  const int hi = lane >> 4, lo = lane & 15;
  const int obase = wave * 64;
  const int pm = tid >> 3;       // P row this thread fills
  const int pj = (tid & 7) * 4;  // P col quad this thread fills

  const float* gbias = bias + ((size_t)bb * N_ + i0 + pm) * N_ + pj;
  float* lbias[2] = {&bias_lds[0][pm * 32 + pj], &bias_lds[1][pm * 32 + pj]};

  v8f acc[4] = {};
  for (int h = 0; h < H_; ++h) {
    const int hb = h * B_ + bb;
    const float* srow = s + (size_t)hb * N_;
    if (tid < 16) {
      row_s[tid] = srow[i0 + tid];
      row_m[tid] = rmax[(size_t)hb * N_ + i0 + tid];
      row_il[tid] = rinv[(size_t)hb * N_ + i0 + tid];
    }
    __syncthreads();
    const float psi = row_s[pm], pmm = row_m[pm], pil = row_il[pm];

    const _Float16* tbase[4];
#pragma unroll
    for (int t = 0; t < 4; ++t)
      tbase[t] = t16t + ((size_t)(hb * FOUT + obase + t * 16 + lo)) * N_ + hi * 8;

    // one tile-body; cur/nxt LDS buffers and the T-fragment set are explicit
    auto body = [&](int jt, AFrag (&bfreg)[4], float* bcur, float* bnxt) {
      const bool more = (jt + 1 < NT_);
      if (more) bias_prefetch(gbias + (size_t)(jt + 1) * 32, bnxt);
      bias_wait(more);  // tile jt resident in bcur
      float p[4];
#pragma unroll
      for (int c = 0; c < 4; ++c) {
        float x = psi + srow[jt * 32 + pj + c];
        x = x > 0.f ? x : LEAKY * x;
        x += bcur[c];
        p[c] = __expf(x - pmm) * pil;
      }
      H2 d0, d1;
      d0.h[0] = (_Float16)p[0]; d0.h[1] = (_Float16)p[1];
      d1.h[0] = (_Float16)p[2]; d1.h[1] = (_Float16)p[3];
      plds[(pm * 32 + pj) >> 1] = d0.u;
      plds[((pm * 32 + pj) >> 1) + 1] = d1.u;
      __syncthreads();
      AFrag a;
      const int abase = (lo * 32 + hi * 8) >> 1;
#pragma unroll
      for (int d = 0; d < 4; ++d) a.u[d] = plds[abase + d];
#pragma unroll
      for (int d = 0; d < 4; ++d) a.u[4 + d] = plds[abase + 8 + d];
#pragma unroll
      for (int t = 0; t < 4; ++t)
        acc[t] = wmma_f16(a.v, bfreg[t].v, acc[t]);
      __syncthreads();
    };

    AFrag bfr0[4], bfr1[4];
    bias_prefetch(gbias, lbias[0]);  // stage bias tile 0
#pragma unroll
    for (int t = 0; t < 4; ++t) load_frag32B(bfr0[t], tbase[t]);  // T tile 0

    for (int jt = 0; jt < NT_; jt += 2) {
      // prefetch T tile jt+1, run even tile
#pragma unroll
      for (int t = 0; t < 4; ++t)
        load_frag32B(bfr1[t], tbase[t] + (size_t)(jt + 1) * 32);
      body(jt, bfr0, lbias[0], lbias[1]);
      // prefetch T tile jt+2, run odd tile
      if (jt + 2 < NT_) {
#pragma unroll
        for (int t = 0; t < 4; ++t)
          load_frag32B(bfr0[t], tbase[t] + (size_t)(jt + 2) * 32);
      }
      body(jt + 1, bfr1, lbias[1], lbias[0]);
    }
  }
  const float scale = 1.f / H_;
#pragma unroll
  for (int t = 0; t < 4; ++t) {
    int o = obase + t * 16 + lo;
#pragma unroll
    for (int r = 0; r < 8; ++r)
      out[((size_t)bb * N_ + i0 + hi * 8 + r) * FOUT + o] = acc[t][r] * scale;
  }
}

// ------------------------------------------------------------------ launch --
extern "C" void kernel_launch(void* const* d_in, const int* in_sizes, int n_in,
                              void* d_out, int out_size, void* d_ws,
                              size_t ws_size, hipStream_t stream) {
  const float* x    = (const float*)d_in[0];  // [B,N,FIN]
  const float* bias = (const float*)d_in[1];  // [B,N,N]
  const float* W    = (const float*)d_in[2];  // [H,FIN,FOUT]
  const float* a    = (const float*)d_in[3];  // [H,FOUT]
  const float* bv   = (const float*)d_in[4];  // [H,FOUT]
  float* out = (float*)d_out;                 // [B,N,FOUT]

  char* ws = (char*)d_ws;
  _Float16* x16  = (_Float16*)ws; ws += (size_t)B_ * N_ * FIN * 2;
  _Float16* w16t = (_Float16*)ws; ws += (size_t)H_ * FOUT * FIN * 2;
  _Float16* t16t = (_Float16*)ws; ws += (size_t)H_ * B_ * FOUT * N_ * 2;
  float* s    = (float*)ws; ws += (size_t)H_ * B_ * N_ * 4;
  float* rmax = (float*)ws; ws += (size_t)H_ * B_ * N_ * 4;
  float* rinv = (float*)ws; ws += (size_t)H_ * B_ * N_ * 4;

  const int nx = B_ * N_ * FIN;
  k_cvt_x<<<(nx + 255) / 256, 256, 0, stream>>>(x, x16, nx);
  const int nw = H_ * FOUT * FIN;
  k_cvt_wt<<<(nw + 255) / 256, 256, 0, stream>>>(W, w16t);
  k_gemm1<<<dim3(N_ / 16, H_ * B_), 128, 0, stream>>>(x16, w16t, bv, t16t);
  k_score<<<dim3(N_ / 256, H_ * B_), 256, 0, stream>>>(t16t, a, s);
  k_stats<<<dim3(N_, B_), 256, 0, stream>>>(bias, s, rmax, rinv);
  k_attn<<<dim3(N_ / 16, B_), 128, 0, stream>>>(bias, s, rmax, rinv, t16t, out);
}
